// AgentAttention_68195490726092
// MI455X (gfx1250) — matrix operand order repaired and checked by reading, not compiled, not run on hardware
//
#include <hip/hip_runtime.h>

// ---------------------------------------------------------------------------
// Agent Attention for MI455X (gfx1250), bf16 WMMA pipeline.
// B=4, N=4096, D=1024, H=16, M=128, DH=64
// Heavy GEMMs use a ping-pong double buffer (manual 2x k-step unroll).
// GEMM kernels are one wave32 per block and carry amdgpu_waves_per_eu(1)
// so the allocator may exceed the 128-VGPR occupancy cap instead of spilling.
// ---------------------------------------------------------------------------

typedef unsigned short u16;
typedef __attribute__((ext_vector_type(16))) __bf16        v16bf;
typedef __attribute__((ext_vector_type(8)))  float         v8f;
typedef __attribute__((ext_vector_type(4)))  unsigned int  v4u;

#define CB 4
#define CN 4096
#define CD 1024
#define CH 16
#define CM 128
#define CDH 64
#define BH (CB*CH)          // 64
#define ROWS (CB*CN)        // 16384
#define SCALE 0.125f        // 64^-0.5

#define GEMM_KERNEL __global__ __launch_bounds__(32) \
    __attribute__((amdgpu_waves_per_eu(1))) void

union Frag { v16bf bf; v4u u[2]; };

__device__ __forceinline__ u16 f2bf(float f) {
    unsigned u = __float_as_uint(f);
    return (u16)((u + 0x7FFFu + ((u >> 16) & 1u)) >> 16);
}
__device__ __forceinline__ float bf2f(u16 h) {
    return __uint_as_float(((unsigned)h) << 16);
}

// A fragment: 16x32 bf16 tile, row-major source, lda in elements.
// Lane l (0..31): row = l&15, K-half = l>>4.  Two contiguous 16B loads.
__device__ __forceinline__ v16bf load_a(const u16* src, int lda, int lane) {
    int m = lane & 15, half = lane >> 4;
    const u16* p = src + (size_t)m * lda + half * 8;
    Frag f;
    f.u[0] = *(const v4u*)(p);
    f.u[1] = *(const v4u*)(p + 16);
    return f.bf;
}
// B fragment from B^T (N x K row-major): 32x16 bf16 tile.
// Lane l: col n = l&15, K-half = l>>4.  Two contiguous 16B loads (k 8h..8h+7, 16+8h..16+8h+7).
__device__ __forceinline__ v16bf load_bt(const u16* srcT, int ldb, int lane) {
    int n = lane & 15, half = lane >> 4;
    const u16* p = srcT + (size_t)n * ldb + half * 8;
    Frag f;
    f.u[0] = *(const v4u*)(p);
    f.u[1] = *(const v4u*)(p + 16);
    return f.bf;
}
__device__ __forceinline__ v8f wmma_bf16(v16bf a, v16bf b, v8f c) {
    return __builtin_amdgcn_wmma_f32_16x16x32_bf16(false, a, false, b,
                                                   (short)0, c, false, false);
}

// Ping-pong pipelined 16x64 C-tile GEMM body: A (16 x K), Bt (64 rows x K).
// Requires K a multiple of 64 (i.e. an even number of 32-wide k-steps).
__device__ __forceinline__ void gemm_16x64_pipelined(
        const u16* __restrict__ A, int lda,
        const u16* __restrict__ Bt, int ldb,
        int K, int lane, v8f acc[4]) {
    v16bf afA = load_a(A, lda, lane);
    v16bf bfA[4];
#pragma unroll
    for (int t = 0; t < 4; ++t)
        bfA[t] = load_bt(Bt + (size_t)(t * 16) * ldb, ldb, lane);
    v16bf afB = load_a(A + 32, lda, lane);
    v16bf bfB[4];
#pragma unroll
    for (int t = 0; t < 4; ++t)
        bfB[t] = load_bt(Bt + (size_t)(t * 16) * ldb + 32, ldb, lane);
#pragma clang loop unroll(disable)
    for (int k0 = 64; k0 < K; k0 += 64) {
        // consume buffer A (step k0-64), refill A with step k0
#pragma unroll
        for (int t = 0; t < 4; ++t)
            acc[t] = wmma_bf16(afA, bfA[t], acc[t]);
        afA = load_a(A + k0, lda, lane);
#pragma unroll
        for (int t = 0; t < 4; ++t)
            bfA[t] = load_bt(Bt + (size_t)(t * 16) * ldb + k0, ldb, lane);
        // consume buffer B (step k0-32), refill B with step k0+32
#pragma unroll
        for (int t = 0; t < 4; ++t)
            acc[t] = wmma_bf16(afB, bfB[t], acc[t]);
        afB = load_a(A + k0 + 32, lda, lane);
#pragma unroll
        for (int t = 0; t < 4; ++t)
            bfB[t] = load_bt(Bt + (size_t)(t * 16) * ldb + k0 + 32, ldb, lane);
    }
#pragma unroll
    for (int t = 0; t < 4; ++t)
        acc[t] = wmma_bf16(afA, bfA[t], acc[t]);
#pragma unroll
    for (int t = 0; t < 4; ++t)
        acc[t] = wmma_bf16(afB, bfB[t], acc[t]);
}

// ---------------------------------------------------------------------------
// Prep / cast kernels
// ---------------------------------------------------------------------------
__global__ void cast_x_kernel(const float* __restrict__ x, u16* __restrict__ xb, int n) {
    int i = blockIdx.x * 256 + threadIdx.x;
    if (i < n) xb[i] = f2bf(x[i]);
}
__global__ void prep_wqkv_kernel(const float* __restrict__ W, u16* __restrict__ Wt) {
    int i = blockIdx.x * 256 + threadIdx.x;          // i = o*1024 + d
    if (i >= 3072 * 1024) return;
    int o = i >> 10, d = i & 1023;
    Wt[i] = f2bf(W[(size_t)d * 3072 + o]);
}
__global__ void prep_wout_kernel(const float* __restrict__ W, u16* __restrict__ Wt) {
    int i = blockIdx.x * 256 + threadIdx.x;          // i = o*1024 + d
    if (i >= 1024 * 1024) return;
    int o = i >> 10, d = i & 1023;
    Wt[i] = f2bf(W[(size_t)d * 1024 + o]);
}
__global__ void prep_agents_kernel(const float* __restrict__ a, u16* __restrict__ aS) {
    int i = blockIdx.x * 256 + threadIdx.x;
    if (i < CH * CM * CDH) aS[i] = f2bf(a[i] * SCALE);
}

// ---------------------------------------------------------------------------
// QKV GEMM: (16384 x 1024) @ WqkvT -> scatter q/k (B,H,N,DH) and vT (B,H,DH,N)
// grid (48, 1024), block 32 (one wave, 16x64 C tile), K=1024 pipelined
// ---------------------------------------------------------------------------
GEMM_KERNEL qkv_gemm_kernel(const u16* __restrict__ xb, const u16* __restrict__ WqkvT,
                            u16* __restrict__ q, u16* __restrict__ k,
                            u16* __restrict__ vT) {
    int lane = threadIdx.x, nh = lane & 15, half = lane >> 4;
    int m0 = blockIdx.y * 16;
    int n0 = blockIdx.x * 64;
    const u16* A  = xb    + (size_t)m0 * 1024;
    const u16* Bt = WqkvT + (size_t)n0 * 1024;
    v8f z = {}; v8f acc[4];
#pragma unroll
    for (int t = 0; t < 4; ++t) acc[t] = z;
    gemm_16x64_pipelined(A, 1024, Bt, 1024, 1024, lane, acc);
#pragma unroll
    for (int t = 0; t < 4; ++t)
#pragma unroll
        for (int r = 0; r < 8; ++r) {
            int token = m0 + half * 8 + r;
            int o = n0 + t * 16 + nh;
            int sel = o >> 10, rem = o & 1023;
            int h = rem >> 6, dh = rem & 63;
            int b = token >> 12, nn = token & 4095;
            size_t bh = (size_t)b * CH + h;
            u16 bv = f2bf(acc[t][r]);
            if (sel == 0)      q [(bh * CN + nn) * CDH + dh] = bv;
            else if (sel == 1) k [(bh * CN + nn) * CDH + dh] = bv;
            else               vT[(bh * CDH + dh) * CN + nn] = bv;
        }
}

// ---------------------------------------------------------------------------
// qa_sim = q @ a^T : per (bh, 16-row tile) -> (16 x 128) f32 scores
// grid (256, 64), block 32.  K=64 (B operand is L1-resident, 16KB)
// ---------------------------------------------------------------------------
GEMM_KERNEL qa_sim_kernel(const u16* __restrict__ q, const u16* __restrict__ aS,
                          float* __restrict__ sim) {
    int lane = threadIdx.x, nh = lane & 15, half = lane >> 4;
    int bh = blockIdx.y;
    int h  = bh & 15;
    int m0 = blockIdx.x * 16;
    const u16* A  = q  + ((size_t)bh * CN + m0) * CDH;
    const u16* Bt = aS + (size_t)h * CM * CDH;          // (M x DH) = B^T
    v8f z = {}; v8f acc[8];
#pragma unroll
    for (int t = 0; t < 8; ++t) acc[t] = z;
#pragma unroll
    for (int k0 = 0; k0 < CDH; k0 += 32) {
        v16bf af = load_a(A + k0, CDH, lane);
#pragma unroll
        for (int t = 0; t < 8; ++t) {
            v16bf bf = load_bt(Bt + (size_t)(t * 16) * CDH + k0, CDH, lane);
            acc[t] = wmma_bf16(af, bf, acc[t]);
        }
    }
#pragma unroll
    for (int t = 0; t < 8; ++t)
#pragma unroll
        for (int r = 0; r < 8; ++r) {
            int row = m0 + half * 8 + r;
            sim[((size_t)bh * CN + row) * CM + t * 16 + nh] = acc[t][r];
        }
}

// ---------------------------------------------------------------------------
// ak_sim = a @ k^T : per (bh, agent tile, 64-col chunk).  K=64
// grid (64, 512), block 32   (y = bh*8 + mtile)
// ---------------------------------------------------------------------------
GEMM_KERNEL ak_sim_kernel(const u16* __restrict__ aS, const u16* __restrict__ k,
                          float* __restrict__ sim) {
    int lane = threadIdx.x, nh = lane & 15, half = lane >> 4;
    int bh = blockIdx.y >> 3;
    int mt = blockIdx.y & 7;
    int h  = bh & 15;
    int c0 = blockIdx.x * 64;
    const u16* A  = aS + ((size_t)h * CM + mt * 16) * CDH;
    const u16* Bt = k  + ((size_t)bh * CN + c0) * CDH;   // keys (N x DH) = B^T
    v8f z = {}; v8f acc[4];
#pragma unroll
    for (int t = 0; t < 4; ++t) acc[t] = z;
#pragma unroll
    for (int k0 = 0; k0 < CDH; k0 += 32) {
        v16bf af = load_a(A + k0, CDH, lane);
#pragma unroll
        for (int t = 0; t < 4; ++t) {
            v16bf bf = load_bt(Bt + (size_t)(t * 16) * CDH + k0, CDH, lane);
            acc[t] = wmma_bf16(af, bf, acc[t]);
        }
    }
#pragma unroll
    for (int t = 0; t < 4; ++t)
#pragma unroll
        for (int r = 0; r < 8; ++r) {
            int m = mt * 16 + half * 8 + r;
            int c = c0 + t * 16 + nh;
            sim[((size_t)bh * CM + m) * CN + c] = acc[t][r];
        }
}

// ---------------------------------------------------------------------------
// Row softmax f32 -> bf16, optional key mask (pre-softmax, like reference)
// grid = rows, block 256
// ---------------------------------------------------------------------------
__global__ void softmax_kernel(const float* __restrict__ in, u16* __restrict__ out,
                               int len, const unsigned char* __restrict__ mask,
                               int rowsPerB, int ncols) {
    int row = blockIdx.x, tid = threadIdx.x;
    const float* ip = in  + (size_t)row * len;
    u16*         op = out + (size_t)row * len;
    const unsigned char* mrow =
        mask ? (mask + (size_t)(row / rowsPerB) * ncols) : nullptr;
    __shared__ float sm[256];
    float lmax = -3.402823e38f;
    for (int j = tid; j < len; j += 256) {
        float s = ip[j];
        if (mrow && !mrow[j]) s = -3.402823e38f;
        lmax = fmaxf(lmax, s);
    }
    sm[tid] = lmax; __syncthreads();
    for (int s = 128; s > 0; s >>= 1) {
        if (tid < s) sm[tid] = fmaxf(sm[tid], sm[tid + s]);
        __syncthreads();
    }
    float rmax = sm[0]; __syncthreads();
    float lsum = 0.f;
    for (int j = tid; j < len; j += 256) {
        float s = ip[j];
        if (mrow && !mrow[j]) s = -3.402823e38f;
        lsum += __expf(s - rmax);
    }
    sm[tid] = lsum; __syncthreads();
    for (int s = 128; s > 0; s >>= 1) {
        if (tid < s) sm[tid] += sm[tid + s];
        __syncthreads();
    }
    float inv = 1.0f / sm[0];
    for (int j = tid; j < len; j += 256) {
        float s = ip[j];
        if (mrow && !mrow[j]) s = -3.402823e38f;
        op[j] = f2bf(__expf(s - rmax) * inv);
    }
}

// ---------------------------------------------------------------------------
// Talking heads: out(b,g,rc) = sum_h W[g,h] * in(b,h,rc)
// ---------------------------------------------------------------------------
__global__ void mix_heads_kernel(const u16* __restrict__ in, const float* __restrict__ W,
                                 u16* __restrict__ out, long RC, long total) {
    __shared__ float w[256];
    int tid = threadIdx.x;
    w[tid] = W[tid];
    __syncthreads();
    long gid = (long)blockIdx.x * 256 + tid;
    if (gid >= total) return;
    long b = gid / RC, rc = gid % RC;
    const u16* ip = in  + (size_t)b * CH * RC + rc;
    u16*       op = out + (size_t)b * CH * RC + rc;
    float xv[CH];
#pragma unroll
    for (int h = 0; h < CH; ++h) xv[h] = bf2f(ip[(size_t)h * RC]);
#pragma unroll
    for (int g = 0; g < CH; ++g) {
        float acc = 0.f;
#pragma unroll
        for (int h = 0; h < CH; ++h) acc += w[g * CH + h] * xv[h];
        op[(size_t)g * RC] = f2bf(acc);
    }
}

// ---------------------------------------------------------------------------
// agent_gathered = ak_attn2 (M x N) @ v (N x DH) ; K = 4096 pipelined
// grid (8, 64), block 32.  Writes f32 output #2 and bf16 transposed copy.
// ---------------------------------------------------------------------------
GEMM_KERNEL ag_gemm_kernel(const u16* __restrict__ ak2, const u16* __restrict__ vT,
                           float* __restrict__ ag_out, u16* __restrict__ agT) {
    int lane = threadIdx.x, nh = lane & 15, half = lane >> 4;
    int bh = blockIdx.y;
    int mt = blockIdx.x;
    const u16* A  = ak2 + ((size_t)bh * CM + mt * 16) * CN;
    const u16* Bt = vT  + (size_t)bh * CDH * CN;        // (DH x N) = B^T
    v8f z = {}; v8f acc[4];
#pragma unroll
    for (int t = 0; t < 4; ++t) acc[t] = z;
    gemm_16x64_pipelined(A, CN, Bt, CN, CN, lane, acc);
#pragma unroll
    for (int t = 0; t < 4; ++t)
#pragma unroll
        for (int r = 0; r < 8; ++r) {
            int m  = mt * 16 + half * 8 + r;
            int dh = t * 16 + nh;
            float v = acc[t][r];
            ag_out[((size_t)bh * CM + m) * CDH + dh] = v;
            agT   [((size_t)bh * CDH + dh) * CM + m] = f2bf(v);
        }
}

// ---------------------------------------------------------------------------
// gates = sigmoid(x @ W_gate + b_gate) : one 128-thread block per token
// ---------------------------------------------------------------------------
__global__ void gates_kernel(const float* __restrict__ x, const float* __restrict__ Wg,
                             const float* __restrict__ bg, float* __restrict__ gates) {
    int row = blockIdx.x, tid = threadIdx.x;
    __shared__ float xs[1024];
    __shared__ float red[128];
    for (int j = tid; j < 1024; j += 128) xs[j] = x[(size_t)row * 1024 + j];
    __syncthreads();
    int h = tid & 15, slice = tid >> 4;   // 8 slices of 128
    float p = 0.f;
    for (int d = slice * 128; d < slice * 128 + 128; ++d)
        p += xs[d] * Wg[(size_t)d * CH + h];
    red[tid] = p; __syncthreads();
    for (int s = 64; s >= 16; s >>= 1) {
        if (tid < s) red[tid] += red[tid + s];
        __syncthreads();
    }
    if (tid < 16) {
        float v = red[tid] + bg[tid];
        gates[(size_t)row * CH + tid] = 1.0f / (1.0f + __expf(-v));
    }
}

// ---------------------------------------------------------------------------
// out = qa_attn2 (N x M) @ agent_gathered (M x DH); gate + mask; -> y bf16
// grid (256, 64), block 32.  K=128 pipelined
// ---------------------------------------------------------------------------
GEMM_KERNEL out_gemm_kernel(const u16* __restrict__ qa2, const u16* __restrict__ agT,
                            const float* __restrict__ gates,
                            const unsigned char* __restrict__ mask,
                            u16* __restrict__ ybf) {
    int lane = threadIdx.x, nh = lane & 15, half = lane >> 4;
    int bh = blockIdx.y;
    int nt = blockIdx.x;
    int b = bh >> 4, h = bh & 15;
    const u16* A  = qa2 + ((size_t)bh * CN + nt * 16) * CM;
    const u16* Bt = agT + (size_t)bh * CDH * CM;        // (DH x M) = B^T
    v8f z = {}; v8f acc[4];
#pragma unroll
    for (int t = 0; t < 4; ++t) acc[t] = z;
    gemm_16x64_pipelined(A, CM, Bt, CM, CM, lane, acc);
#pragma unroll
    for (int t = 0; t < 4; ++t)
#pragma unroll
        for (int r = 0; r < 8; ++r) {
            int token = nt * 16 + half * 8 + r;
            int dh = t * 16 + nh;
            size_t tk = (size_t)b * CN + token;
            float g = gates[tk * CH + h];
            float v = acc[t][r] * g;
            if (!mask[tk]) v = 0.f;
            ybf[tk * CD + h * CDH + dh] = f2bf(v);
        }
}

// ---------------------------------------------------------------------------
// final = y (16384 x 1024) @ W_out -> f32 output #1.  K=1024 pipelined
// grid (16, 1024), block 32
// ---------------------------------------------------------------------------
GEMM_KERNEL final_gemm_kernel(const u16* __restrict__ ybf, const u16* __restrict__ WoutT,
                              float* __restrict__ out) {
    int lane = threadIdx.x, nh = lane & 15, half = lane >> 4;
    int m0 = blockIdx.y * 16;
    int n0 = blockIdx.x * 64;
    const u16* A  = ybf   + (size_t)m0 * 1024;
    const u16* Bt = WoutT + (size_t)n0 * 1024;
    v8f z = {}; v8f acc[4];
#pragma unroll
    for (int t = 0; t < 4; ++t) acc[t] = z;
    gemm_16x64_pipelined(A, 1024, Bt, 1024, 1024, lane, acc);
#pragma unroll
    for (int t = 0; t < 4; ++t)
#pragma unroll
        for (int r = 0; r < 8; ++r) {
            int token = m0 + half * 8 + r;
            int o = n0 + t * 16 + nh;
            out[(size_t)token * 1024 + o] = acc[t][r];
        }
}

// ---------------------------------------------------------------------------
extern "C" void kernel_launch(void* const* d_in, const int* in_sizes, int n_in,
                              void* d_out, int out_size, void* d_ws, size_t ws_size,
                              hipStream_t stream) {
    (void)in_sizes; (void)n_in; (void)out_size; (void)ws_size;
    const float* x      = (const float*)d_in[0];
    const float* W_qkv  = (const float*)d_in[1];
    const float* W_gate = (const float*)d_in[2];
    const float* b_gate = (const float*)d_in[3];
    const float* agents = (const float*)d_in[4];
    const float* W_qa   = (const float*)d_in[5];
    const float* W_ak   = (const float*)d_in[6];
    const float* W_out  = (const float*)d_in[7];
    const unsigned char* mask = (const unsigned char*)d_in[8];

    float* out1 = (float*)d_out;                    // (B,N,D)
    float* out2 = out1 + (size_t)ROWS * CD;         // (B,H,M,DH)

    char* w = (char*)d_ws;
    u16* wqkvT = (u16*)w;  w += (size_t)3072 * 1024 * 2;
    u16* woutT = (u16*)w;  w += (size_t)1024 * 1024 * 2;
    u16* aS    = (u16*)w;  w += (size_t)CH * CM * CDH * 2;
    u16* xbf   = (u16*)w;  w += (size_t)ROWS * CD * 2;
    u16* qb    = (u16*)w;  w += (size_t)BH * CN * CDH * 2;
    u16* kb    = (u16*)w;  w += (size_t)BH * CN * CDH * 2;
    u16* vT    = (u16*)w;  w += (size_t)BH * CDH * CN * 2;
    float* sim = (float*)w; w += (size_t)BH * CN * CM * 4;
    u16* attnA = (u16*)w;  w += (size_t)BH * CN * CM * 2;   // raw qa_attn, then ak_attn
    u16* qa2   = (u16*)w;  w += (size_t)BH * CN * CM * 2;
    u16* ak2   = (u16*)w;  w += (size_t)BH * CM * CN * 2;
    u16* agT   = (u16*)w;  w += (size_t)BH * CDH * CM * 2;
    float* gts = (float*)w; w += (size_t)ROWS * CH * 4;
    u16* ybf   = (u16*)w;  w += (size_t)ROWS * CD * 2;

    // --- prep ---
    cast_x_kernel<<<(ROWS * CD + 255) / 256, 256, 0, stream>>>(x, xbf, ROWS * CD);
    prep_wqkv_kernel<<<(3072 * 1024 + 255) / 256, 256, 0, stream>>>(W_qkv, wqkvT);
    prep_wout_kernel<<<(1024 * 1024 + 255) / 256, 256, 0, stream>>>(W_out, woutT);
    prep_agents_kernel<<<(CH * CM * CDH + 255) / 256, 256, 0, stream>>>(agents, aS);

    // --- QKV projection ---
    qkv_gemm_kernel<<<dim3(48, ROWS / 16), 32, 0, stream>>>(xbf, wqkvT, qb, kb, vT);

    // --- qa path: sim -> softmax -> talking heads ---
    qa_sim_kernel<<<dim3(CN / 16, BH), 32, 0, stream>>>(qb, aS, sim);
    softmax_kernel<<<BH * CN, 256, 0, stream>>>(sim, attnA, CM, nullptr, 1, 0);
    mix_heads_kernel<<<(CB * CN * CM + 255) / 256, 256, 0, stream>>>(
        attnA, W_qa, qa2, (long)CN * CM, (long)CB * CN * CM);

    // --- ak path: sim -> masked softmax -> talking heads ---
    ak_sim_kernel<<<dim3(CN / 64, BH * 8), 32, 0, stream>>>(aS, kb, sim);
    softmax_kernel<<<BH * CM, 256, 0, stream>>>(sim, attnA, CN, mask, CH * CM, CN);
    mix_heads_kernel<<<(CB * CM * CN + 255) / 256, 256, 0, stream>>>(
        attnA, W_ak, ak2, (long)CM * CN, (long)CB * CM * CN);

    // --- agent_gathered (output #2) ---
    ag_gemm_kernel<<<dim3(CM / 16, BH), 32, 0, stream>>>(ak2, vT, out2, agT);

    // --- gates, gated output, final projection ---
    gates_kernel<<<ROWS, 128, 0, stream>>>(x, W_gate, b_gate, gts);
    out_gemm_kernel<<<dim3(CN / 16, BH), 32, 0, stream>>>(qa2, agT, gts, mask, ybf);
    final_gemm_kernel<<<dim3(16, ROWS / 16), 32, 0, stream>>>(ybf, woutT, out1);
}
